// MYLSTM1_15831249453425
// MI455X (gfx1250) — matrix operand order repaired and checked by reading, not compiled, not run on hardware
//
#include <hip/hip_runtime.h>
#include <stdint.h>

// Problem constants (match reference)
#define BB  32
#define TT  4096
#define DD  256
#define UU  256
#define N4U 1024   // 4*U fused gate columns (permuted: p = 4*u + gate)
#define KH  256    // K rows of W multiplying the carried state (cat = [s, x])
#define PACKW_HALVES (N4U * KH)   // 262144 halves per packed weight matrix

#if defined(__has_builtin)
#if __has_builtin(__builtin_amdgcn_global_load_async_to_lds_b128) && \
    __has_builtin(__builtin_amdgcn_s_wait_asynccnt)
#define HAVE_ASYNC_LDS 1
#endif
#endif

typedef __attribute__((ext_vector_type(16))) __bf16 v16bf;
typedef __attribute__((ext_vector_type(8)))  float  v8f;

struct Frag32B { uint4 lo; uint4 hi; };

__device__ __forceinline__ uint16_t f32_bf16(float x) {
  uint32_t u = __float_as_uint(x);
  u += 0x7FFFu + ((u >> 16) & 1u);      // round-to-nearest-even
  return (uint16_t)(u >> 16);
}
__device__ __forceinline__ float bf16_f32(uint16_t h) {
  return __uint_as_float(((uint32_t)h) << 16);
}
__device__ __forceinline__ v8f vzero8() {
  v8f v = {0.f, 0.f, 0.f, 0.f, 0.f, 0.f, 0.f, 0.f};
  return v;
}
__device__ __forceinline__ float sigm(float x) { return 1.f / (1.f + __expf(-x)); }

#if HAVE_ASYNC_LDS
// The builtin's params are pointers to 16-byte int vectors in AS1 / AS3
// (per hipcc diagnostic: "__vector_size__(4 * sizeof(int)) int __device__ *").
typedef int v4i_b __attribute__((vector_size(16)));
typedef __attribute__((address_space(1))) v4i_b* gvec_t;
typedef __attribute__((address_space(3))) v4i_b* lvec_t;
// 16-byte global->LDS async copy (ASYNCcnt-tracked). Generic LDS pointer's low
// 32 bits are the LDS offset (ISA 10.2 aperture rules), so integer casts give
// us well-formed AS1/AS3 pointers without addrspacecasts.
__device__ __forceinline__ void async_copy16(const void* g, void* l) {
  __builtin_amdgcn_global_load_async_to_lds_b128(
      (gvec_t)(uintptr_t)g, (lvec_t)(uint32_t)(uintptr_t)l, 0, 0);
}
__device__ __forceinline__ void async_wait0() {
  __builtin_amdgcn_s_wait_asynccnt(0);
}
#endif

// Bulk copy of `n16` 16-byte chunks from global to LDS, strided by 256 threads.
__device__ __forceinline__ void stage_to_lds(const uint4* __restrict__ src,
                                             uint4* __restrict__ dst,
                                             int n16, int tid) {
#if HAVE_ASYNC_LDS
  for (int i = tid; i < n16; i += 256) async_copy16(src + i, dst + i);
  async_wait0();
#else
  for (int i = tid; i < n16; i += 256) dst[i] = src[i];
#endif
}

// --- WMMA fragment loaders (bf16 16x16x32, layouts per CDNA5 ISA 7.12.2) ---
// A (16x32, MxK): lane<16 holds K[0..7],K[16..23]; lane>=16 holds K[8..15],K[24..31]
__device__ __forceinline__ v16bf load_a_frag(const uint16_t* a, int strideHalves,
                                             int mbase, int kbase, int lane) {
  int row = mbase + (lane & 15);
  int c   = (lane >> 4) * 8;
  const uint16_t* p = a + (size_t)row * strideHalves + kbase + c;
  Frag32B f;
  f.lo = *(const uint4*)(p);        // K c..c+7
  f.hi = *(const uint4*)(p + 16);   // K c+16..c+23
  return __builtin_bit_cast(v16bf, f);
}
// B (32x16, KxN) pre-packed frag-major: lane holds 16 contiguous halves
// (lane&15 = N column; lane>>4 selects K 0..7,16..23 vs 8..15,24..31)
__device__ __forceinline__ v16bf load_b_frag(const uint16_t* packed, int fragIdx, int lane) {
  const uint16_t* p = packed + ((size_t)fragIdx * 32 + lane) * 16;
  Frag32B f;
  f.lo = *(const uint4*)(p);
  f.hi = *(const uint4*)(p + 8);
  return __builtin_bit_cast(v16bf, f);
}

// ---------------------------------------------------------------------------
// Kernel 1: pack Wf/Wi/Wc/Wo (f32, (512,256) each) into bf16 WMMA-B-fragment
// order, with gate-interleaved column permutation p = 4*u + g.
// packWh = rows 0..255 (state part), packWx = rows 256..511 (x part).
// Packed index: (((nt*8 + kc)*32 + lane)*16 + j)
// ---------------------------------------------------------------------------
__global__ void __launch_bounds__(256) lstm_pack_w(
    const float* __restrict__ Wf, const float* __restrict__ Wi,
    const float* __restrict__ Wc, const float* __restrict__ Wo,
    uint16_t* __restrict__ packWh, uint16_t* __restrict__ packWx) {
  int idx  = blockIdx.x * 256 + threadIdx.x;  // 0 .. 262143
  int j    = idx & 15;
  int lane = (idx >> 4) & 31;
  int kc   = (idx >> 9) & 7;
  int nt   = idx >> 12;                       // 0..63
  int p    = nt * 16 + (lane & 15);           // permuted column
  int u    = p >> 2;
  int g    = p & 3;
  int k    = kc * 32 + (lane >> 4) * 8 + (j & 7) + (j >> 3) * 16;  // 0..255
  const float* W = (g == 0) ? Wf : (g == 1) ? Wi : (g == 2) ? Wc : Wo;
  packWh[idx] = f32_bf16(W[(size_t)k * UU + u]);          // state rows come first
  packWx[idx] = f32_bf16(W[(size_t)(KH + k) * UU + u]);   // x rows
}

// ---------------------------------------------------------------------------
// Kernel 2: Zx[t][b][p] = bf16( x[b,t,:] @ Wx[:,p] )   (fully parallel GEMM)
// grid (4, T/16); block 256 (8 waves). Each block: one 256-col chunk, 16 ts.
// Wx chunk staged in LDS once (async), reused across 16 timesteps.
// ---------------------------------------------------------------------------
__global__ void __launch_bounds__(256) lstm_zx_gemm(
    const float* __restrict__ x, const uint16_t* __restrict__ packWx,
    uint16_t* __restrict__ Zx) {
  extern __shared__ uint8_t smem[];
  uint16_t* wxT = (uint16_t*)smem;             // 16 nt * 4096 halves = 128 KB
  uint16_t* xA  = (uint16_t*)(smem + 131072);  // 32x256 bf16 = 16 KB
  const int tid  = threadIdx.x;
  const int lane = tid & 31;
  const int w    = tid >> 5;
  const int ug4  = blockIdx.x;                 // 0..3: columns [ug4*256, +256)
  const int t0   = blockIdx.y * 16;

  stage_to_lds((const uint4*)(packWx + (size_t)ug4 * 16 * 4096),
               (uint4*)wxT, 8192, tid);
  __syncthreads();

  const int m   = (w & 1);    // M tile: batches m*16 ..
  const int ntg = (w >> 1);   // 4 N tiles: ntL in [ntg*4, ntg*4+4)

  for (int tt = 0; tt < 16; ++tt) {
    int t = t0 + tt;
    for (int i = tid; i < BB * DD; i += 256) {   // stage x[:, t, :] as bf16
      int b = i >> 8, k = i & 255;
      xA[i] = f32_bf16(x[((size_t)b * TT + t) * DD + k]);
    }
    __syncthreads();

    v16bf aF[8];
#pragma unroll
    for (int kc = 0; kc < 8; ++kc)
      aF[kc] = load_a_frag(xA, DD, m * 16, kc * 32, lane);

    v8f acc[4];
#pragma unroll
    for (int q = 0; q < 4; ++q) acc[q] = vzero8();
#pragma unroll
    for (int kc = 0; kc < 8; ++kc) {
#pragma unroll
      for (int q = 0; q < 4; ++q) {
        int ntL = ntg * 4 + q;
        v16bf bf = load_b_frag(wxT, ntL * 8 + kc, lane);
        acc[q] = __builtin_amdgcn_wmma_f32_16x16x32_bf16(
            false, aF[kc], false, bf, (short)0, acc[q], false, false);
      }
    }
    // C layout: VGPR r -> M = r + 8*(lane>=16); N = lane&15
#pragma unroll
    for (int q = 0; q < 4; ++q) {
      int ntL  = ntg * 4 + q;
      int pcol = ug4 * 256 + ntL * 16 + (lane & 15);
#pragma unroll
      for (int r = 0; r < 8; ++r) {
        int b = m * 16 + r + ((lane >> 4) << 3);
        Zx[((size_t)t * BB + b) * N4U + pcol] = f32_bf16(acc[q][r]);
      }
    }
    __syncthreads();
  }
}

// ---------------------------------------------------------------------------
// Kernel 3: sequential recurrence. grid (2,2): x = unit-group, y = batch-group.
// Each WG: z = S(16x256) @ Wh(256x512 slice, LDS-resident) via WMMA,
// gate math in f32, ct exchanged with the peer unit-group WG through L2
// (monotonic atomic counter barrier, double-buffered by step parity).
// ---------------------------------------------------------------------------
__global__ void __launch_bounds__(256) lstm_recurrent(
    const float* __restrict__ h0, const uint16_t* __restrict__ packWh,
    const uint16_t* __restrict__ Zx, uint16_t* __restrict__ Sex,
    unsigned int* __restrict__ counters, float* __restrict__ out) {
  extern __shared__ uint8_t smem[];
  uint16_t* whT  = (uint16_t*)smem;                          // 256 KB
  uint16_t* sA   = (uint16_t*)(smem + 262144);               // 16x256 bf16, 8 KB
  float*    zbuf = (float*)(smem + 262144 + 8192);           // 16x512 f32, 32 KB
  float*    sOld = (float*)(smem + 262144 + 8192 + 32768);   // 16x128 f32, 8 KB
  const int tid = threadIdx.x, lane = tid & 31, w = tid >> 5;
  const int ug = blockIdx.x, bg = blockIdx.y;
  const int pug = 1 - ug;

  // stage Wh slice: nt in [ug*32, ug*32+32), frag layout preserved
  stage_to_lds((const uint4*)(packWh + (size_t)ug * 32 * 4096),
               (uint4*)whT, 16384, tid);
  for (int i = tid; i < 16 * 256; i += 256) {   // initial state (full K)
    int bl = i >> 8, k = i & 255;
    sA[i] = f32_bf16(h0[(size_t)(bg * 16 + bl) * UU + k]);
  }
  for (int i = tid; i < 16 * 128; i += 256) {   // carried cell, f32, own slice
    int bl = i >> 7, ul = i & 127;
    sOld[i] = h0[(size_t)(bg * 16 + bl) * UU + ug * 128 + ul];
  }
  __syncthreads();

  unsigned int* cnt = counters + bg;

  for (int t = 0; t < TT; ++t) {
    // ---- recurrent GEMM: each wave owns 4 N-tiles (64 permuted cols) ----
    v16bf aF[8];
#pragma unroll
    for (int kc = 0; kc < 8; ++kc)
      aF[kc] = load_a_frag(sA, 256, 0, kc * 32, lane);

    v8f acc[4];
#pragma unroll
    for (int q = 0; q < 4; ++q) acc[q] = vzero8();
#pragma unroll
    for (int kc = 0; kc < 8; ++kc) {
#pragma unroll
      for (int q = 0; q < 4; ++q) {
        int ntL = w * 4 + q;
        v16bf bf = load_b_frag(whT, ntL * 8 + kc, lane);
        acc[q] = __builtin_amdgcn_wmma_f32_16x16x32_bf16(
            false, aF[kc], false, bf, (short)0, acc[q], false, false);
      }
    }
#pragma unroll
    for (int q = 0; q < 4; ++q) {
      int col = (w * 4 + q) * 16 + (lane & 15);
#pragma unroll
      for (int r = 0; r < 8; ++r) {
        int row = r + ((lane >> 4) << 3);
        zbuf[row * 512 + col] = acc[q][r];
      }
    }
    __syncthreads();

    // ---- gates: 16 batches x 128 units per WG; 4 gates contiguous (p=4u+g)
    const size_t zxBase = ((size_t)t * BB + bg * 16) * N4U + (size_t)ug * 512;
#pragma unroll
    for (int i = 0; i < 8; ++i) {
      int idx = i * 256 + tid;          // 0..2047
      int ul = idx & 127, bl = idx >> 7;
      int u = ug * 128 + ul, b = bg * 16 + bl;
      const float4 zc = *(const float4*)&zbuf[bl * 512 + 4 * ul];
      uint2 zxp = *(const uint2*)(Zx + zxBase + (size_t)bl * N4U + 4 * ul);
      float zf = zc.x + bf16_f32((uint16_t)(zxp.x & 0xFFFFu));
      float zi = zc.y + bf16_f32((uint16_t)(zxp.x >> 16));
      float zg = zc.z + bf16_f32((uint16_t)(zxp.y & 0xFFFFu));
      float zo = zc.w + bf16_f32((uint16_t)(zxp.y >> 16));
      float fg = sigm(zf), ig = sigm(zi), gg = sigm(zg);
      float c  = ig * gg + fg * sOld[bl * 128 + ul];   // old cell == carried ct
      float o  = tanhf(zo);
      float ct = tanhf(c);
      out[((size_t)b * TT + t) * UU + u]        = o * ct; // H
      out[((size_t)(BB + b) * TT + t) * UU + u] = ct;     // C
      sOld[bl * 128 + ul] = ct;
      uint16_t cb = f32_bf16(ct);
      sA[bl * 256 + u] = cb;                              // own half
      Sex[(size_t)((t + 1) & 1) * (BB * UU) + (size_t)b * UU + u] = cb;
    }
    if (t + 1 < TT) {  // pull next step's Zx slice toward the WGP
      __builtin_prefetch(Zx + ((size_t)(t + 1) * BB + bg * 16) * N4U +
                             (size_t)ug * 512 + (size_t)tid * 8, 0, 1);
    }
    __threadfence();
    __syncthreads();
    if (tid == 0) {  // pairwise step barrier with the peer unit-group WG
      __hip_atomic_fetch_add(cnt, 1u, __ATOMIC_RELEASE, __HIP_MEMORY_SCOPE_AGENT);
      while (__hip_atomic_load(cnt, __ATOMIC_ACQUIRE, __HIP_MEMORY_SCOPE_AGENT) <
             2u * (unsigned)(t + 1)) {
        __builtin_amdgcn_s_sleep(1);
      }
    }
    __syncthreads();
    // per-wave acquire so every CU's L0 is invalidated before reading peer data
    (void)__hip_atomic_load(cnt, __ATOMIC_ACQUIRE, __HIP_MEMORY_SCOPE_AGENT);

    { // pull the peer's half of the new state into sA (256 x 16B chunks)
      const uint16_t* src = Sex + (size_t)((t + 1) & 1) * (BB * UU);
      int c8 = tid;                 // one 8-half chunk per thread
      int bl = c8 >> 4, j = c8 & 15;
      int b = bg * 16 + bl;
      const uint16_t* sp = src + (size_t)b * UU + pug * 128 + j * 8;
      uint16_t* dp = sA + bl * 256 + pug * 128 + j * 8;
#if HAVE_ASYNC_LDS
      async_copy16(sp, dp);
      async_wait0();
#else
      *(uint4*)dp = *(const uint4*)sp;
#endif
    }
    __syncthreads();
  }
}

// ---------------------------------------------------------------------------
// Host launcher. Workspace layout (assumes ws_size >= ~258 MB):
//   [0,256)            step-barrier counters (zeroed each launch)
//   [256, +32KB)       Sex: double-buffered bf16 state exchange [2][32][256]
//   [64KB, +512KB)     packWh (bf16, frag-packed)
//   [64KB+512KB, ...)  packWx (bf16, frag-packed)
//   [1114112, +256MB)  Zx bf16 [T][B][1024]
// ---------------------------------------------------------------------------
extern "C" void kernel_launch(void* const* d_in, const int* in_sizes, int n_in,
                              void* d_out, int out_size, void* d_ws, size_t ws_size,
                              hipStream_t stream) {
  const float* x  = (const float*)d_in[0];
  const float* h0 = (const float*)d_in[1];
  const float* Wf = (const float*)d_in[2];
  const float* Wi = (const float*)d_in[3];
  const float* Wc = (const float*)d_in[4];
  const float* Wo = (const float*)d_in[5];
  float* out = (float*)d_out;

  uint8_t* ws = (uint8_t*)d_ws;
  unsigned int* counters = (unsigned int*)(ws + 0);
  uint16_t* Sex    = (uint16_t*)(ws + 256);
  uint16_t* packWh = (uint16_t*)(ws + (64u << 10));
  uint16_t* packWx = (uint16_t*)(ws + (64u << 10) + (size_t)PACKW_HALVES * 2);
  uint16_t* Zx     = (uint16_t*)(ws + (64u << 10) + (size_t)PACKW_HALVES * 4);

  (void)hipMemsetAsync(counters, 0, 256, stream);

  lstm_pack_w<<<PACKW_HALVES / 256, 256, 0, stream>>>(Wf, Wi, Wc, Wo, packWh, packWx);

  lstm_zx_gemm<<<dim3(4, TT / 16), 256, 131072 + 16384, stream>>>(x, packWx, Zx);

  lstm_recurrent<<<dim3(2, 2), 256, 262144 + 8192 + 32768 + 8192, stream>>>(
      h0, packWh, Zx, Sex, counters, out);
}
